// SAGE2_47004122087520
// MI455X (gfx1250) — matrix operand order repaired and checked by reading, not compiled, not run on hardware
//
#include <hip/hip_runtime.h>
#include <hip/hip_bf16.h>

// ---------------------------------------------------------------------------
// Types for CDNA5 WMMA (wave32): v_wmma_f32_16x16x32_bf16
// ---------------------------------------------------------------------------
typedef __attribute__((ext_vector_type(16))) __bf16 v16bf;
typedef __attribute__((ext_vector_type(8)))  float  v8f;

static __device__ __forceinline__ unsigned short f2bf(float f) {
    unsigned u = __float_as_uint(f);
    unsigned r = (u + 0x7FFFu + ((u >> 16) & 1u)) >> 16;   // RNE
    return (unsigned short)r;
}
static __device__ __forceinline__ float bf2f(unsigned short h) {
    return __uint_as_float(((unsigned)h) << 16);
}

// CDNA5 async direct global->LDS copy (ASYNCcnt path, no VGPR round-trip).
// VDST = wave-relative LDS byte address; VADDR = 64-bit global address.
static __device__ __forceinline__ void async_load_b128(unsigned lds_off, const void* gptr) {
    asm volatile("global_load_async_to_lds_b128 %0, %1, off"
                 :: "v"(lds_off), "v"((unsigned long long)(size_t)gptr)
                 : "memory");
}
// Async loads complete in order: <=4 outstanding => oldest 4 (current tile) done.
static __device__ __forceinline__ void wait_async_le4() {
    asm volatile("s_wait_asynccnt 0x4" ::: "memory");
}
static __device__ __forceinline__ void wait_async_zero() {
    asm volatile("s_wait_asynccnt 0x0" ::: "memory");
}

// ---------------------------------------------------------------------------
// Elementwise helpers
// ---------------------------------------------------------------------------
__global__ void zero_f32_kernel(float* __restrict__ p, long long n) {
    long long i = (long long)blockIdx.x * blockDim.x + threadIdx.x;
    if (i < n) p[i] = 0.0f;
}

__global__ void convert_f32_bf16_kernel(const float* __restrict__ in,
                                        unsigned short* __restrict__ out,
                                        long long n) {
    long long i = (long long)blockIdx.x * blockDim.x + threadIdx.x;
    if (i < n) out[i] = f2bf(in[i]);
}

// WT[n,k] = bf16(W[k,n])  (weights are small; done once)
__global__ void transpose_f32_bf16_kernel(const float* __restrict__ W,
                                          unsigned short* __restrict__ WT,
                                          int K, int N) {
    long long i = (long long)blockIdx.x * blockDim.x + threadIdx.x;
    if (i >= (long long)K * N) return;
    int k = (int)(i / N);
    int n = (int)(i % N);
    WT[(size_t)n * K + k] = f2bf(W[i]);
}

__global__ void degree_kernel(const int* __restrict__ dst, float* __restrict__ deg, int E) {
    int e = blockIdx.x * blockDim.x + threadIdx.x;
    if (e < E) atomicAdd(&deg[dst[e]], 1.0f);
}

// A[dst[e], :] += x[src[e], :]   (f32 source), one wave32 per edge
__global__ void scatter_add_f32_kernel(const float* __restrict__ x,
                                       const int* __restrict__ src,
                                       const int* __restrict__ dst,
                                       float* __restrict__ A, int E, int D) {
    long long gid = (long long)blockIdx.x * blockDim.x + threadIdx.x;
    int e = (int)(gid >> 5);
    int lane = (int)(gid & 31);
    if (e >= E) return;
    const float* xs = x + (size_t)src[e] * D;
    float*       Ad = A + (size_t)dst[e] * D;
    for (int j = lane; j < D; j += 32)
        atomicAdd(&Ad[j], xs[j]);
}

// A[dst[e], :] += float(h_bf16[src[e], :]), one wave32 per edge
__global__ void scatter_add_bf16_kernel(const unsigned short* __restrict__ h,
                                        const int* __restrict__ src,
                                        const int* __restrict__ dst,
                                        float* __restrict__ A, int E, int D) {
    long long gid = (long long)blockIdx.x * blockDim.x + threadIdx.x;
    int e = (int)(gid >> 5);
    int lane = (int)(gid & 31);
    if (e >= E) return;
    const unsigned short* hs = h + (size_t)src[e] * D;
    float*                Ad = A + (size_t)dst[e] * D;
    for (int j = lane; j < D; j += 32)
        atomicAdd(&Ad[j], bf2f(hs[j]));
}

// out_bf16[i,k] = bf16(A[i,k] / max(deg[i], 1))
__global__ void normalize_bf16_kernel(const float* __restrict__ A,
                                      const float* __restrict__ deg,
                                      unsigned short* __restrict__ out,
                                      int Nn, int D) {
    long long i = (long long)blockIdx.x * blockDim.x + threadIdx.x;
    long long n = (long long)Nn * D;
    if (i >= n) return;
    int row = (int)(i / D);
    out[i] = f2bf(A[i] / fmaxf(deg[row], 1.0f));
}

// ---------------------------------------------------------------------------
// Fused dual-source GEMM with double-buffered async pipeline:
//   C = A0 @ B0^T + A1 @ B1^T + bias   (+ optional ReLU)
// A sources: bf16 [M x K] row-major. B sources: pre-transposed bf16 [N x K].
// Tiles staged via global_load_async_to_lds_b128 into ping-pong LDS buffers;
// tile t+1 DMA overlaps tile t WMMA, synchronized with s_wait_asynccnt 4.
// Block tile 64x64, 4 waves, each wave a 32x32 quadrant of 2x2 WMMA tiles.
// ---------------------------------------------------------------------------
#define TM 64
#define TN 64
#define TK 32
#define TSTRIDE 40                 // 32 + 8 halves pad (80B rows, 16B aligned)
#define BUFHALVES (TM * TSTRIDE)   // 2560 halves = 5120 B per tile buffer

template <bool RELU, bool OUT_BF16>
__global__ __launch_bounds__(128) void sage_gemm_kernel(
    const unsigned short* __restrict__ Ag0, const unsigned short* __restrict__ Bt0,
    const unsigned short* __restrict__ Ag1, const unsigned short* __restrict__ Bt1,
    const float* __restrict__ bias, void* __restrict__ Cout,
    int M, int N, int K) {

    __shared__ unsigned short As[2 * BUFHALVES];
    __shared__ unsigned short Bs[2 * BUFHALVES];

    const int tileM = blockIdx.y * TM;
    const int tileN = blockIdx.x * TN;
    const int tid   = threadIdx.x;        // 0..127
    const int wave  = tid >> 5;           // 0..3
    const int lane  = tid & 31;
    const int waveM = (wave >> 1) * 32;
    const int waveN = (wave & 1) * 32;

    const unsigned asBase = (unsigned)(size_t)&As[0];
    const unsigned bsBase = (unsigned)(size_t)&Bs[0];

    v8f acc[2][2];
    #pragma unroll
    for (int i = 0; i < 2; ++i)
        #pragma unroll
        for (int j = 0; j < 2; ++j)
            acc[i][j] = v8f{};

    // Fragment decode per ISA 7.12.2 (wave32, bf16 A 16x32 / B 32x16)
    const int mfrag = lane & 15;
    const int kbA   = (lane >> 4) * 8;
    const int nfrag = lane & 15;
    const int kbB   = (lane >> 4) * 8;

    // per-thread staging slot: 64 rows x 4 chunks of 8 halves, 2 per thread
    const int row0 = tid >> 2;            // 0..31
    const int c8   = tid & 3;             // 0..3
    const unsigned ldsOff0 = (unsigned)((row0     ) * TSTRIDE + c8 * 8) * 2;
    const unsigned ldsOff1 = (unsigned)((row0 + 32) * TSTRIDE + c8 * 8) * 2;

    const int nIter = (2 * K) / TK;       // both sources, flattened

    // issue the 4 async b128 loads for iteration `it` into buffer `buf`
    auto issue = [&](int it, int buf) {
        const int kk = it * TK;
        const unsigned short* Ag;
        const unsigned short* Bt;
        int k0;
        if (kk < K) { Ag = Ag0; Bt = Bt0; k0 = kk; }
        else        { Ag = Ag1; Bt = Bt1; k0 = kk - K; }
        int ga0 = tileM + row0;      if (ga0 >= M) ga0 = M - 1;  // clamp: rows>=M never stored
        int ga1 = tileM + row0 + 32; if (ga1 >= M) ga1 = M - 1;
        const unsigned aB = asBase + (unsigned)buf * (BUFHALVES * 2);
        const unsigned bB = bsBase + (unsigned)buf * (BUFHALVES * 2);
        async_load_b128(aB + ldsOff0, Ag + (size_t)ga0 * K + k0 + c8 * 8);
        async_load_b128(aB + ldsOff1, Ag + (size_t)ga1 * K + k0 + c8 * 8);
        async_load_b128(bB + ldsOff0, Bt + (size_t)(tileN + row0     ) * K + k0 + c8 * 8);
        async_load_b128(bB + ldsOff1, Bt + (size_t)(tileN + row0 + 32) * K + k0 + c8 * 8);
    };

    issue(0, 0);                          // prologue: fill buffer 0

    #pragma unroll 1
    for (int it = 0; it < nIter; ++it) {
        const int buf = it & 1;

        if (it + 1 < nIter) {
            issue(it + 1, buf ^ 1);       // overlap next tile DMA with this tile's WMMA
            wait_async_le4();             // oldest 4 (current tile) complete
        } else {
            wait_async_zero();
        }
        if (it + 2 < nIter) {             // warm L2 two tiles ahead
            const int kk2 = (it + 2) * TK;
            const unsigned short* Ap = (kk2 < K) ? Ag0 : Ag1;
            const unsigned short* Bp = (kk2 < K) ? Bt0 : Bt1;
            const int k2 = (kk2 < K) ? kk2 : kk2 - K;
            __builtin_prefetch(Ap + (size_t)(tileM + row0) * K + k2, 0, 1);
            __builtin_prefetch(Bp + (size_t)(tileN + row0) * K + k2, 0, 1);
        }
        __syncthreads();                  // all waves' current-tile DMA landed

        const unsigned abuf = buf * BUFHALVES;
        const unsigned bbuf = buf * BUFHALVES;

        // --- fragments: contiguous per-lane runs -> ds_load_b128 pairs
        v16bf afrag[2];
        v16bf bfrag[2];
        #pragma unroll
        for (int i = 0; i < 2; ++i) {
            const int mrow = waveM + i * 16 + mfrag;
            #pragma unroll
            for (int q = 0; q < 8; ++q) {
                int kk = kbA + ((q < 4) ? 2 * q : 16 + 2 * (q - 4));
                afrag[i][2 * q]     = __builtin_bit_cast(__bf16, As[abuf + mrow * TSTRIDE + kk]);
                afrag[i][2 * q + 1] = __builtin_bit_cast(__bf16, As[abuf + mrow * TSTRIDE + kk + 1]);
            }
        }
        #pragma unroll
        for (int j = 0; j < 2; ++j) {
            const int nrow = waveN + j * 16 + nfrag;
            #pragma unroll
            for (int q = 0; q < 16; ++q)
                bfrag[j][q] = __builtin_bit_cast(__bf16, Bs[bbuf + nrow * TSTRIDE + kbB + ((q < 8) ? q : q + 8)]);
        }

        // --- 4x v_wmma_f32_16x16x32_bf16; reuse_a hint on A-sharing pairs
        #pragma unroll
        for (int i = 0; i < 2; ++i) {
            acc[i][0] = __builtin_amdgcn_wmma_f32_16x16x32_bf16(
                false, afrag[i], false, bfrag[0], (short)0, acc[i][0], false, false);
            acc[i][1] = __builtin_amdgcn_wmma_f32_16x16x32_bf16(
                false, afrag[i], false, bfrag[1], (short)0, acc[i][1], true, false);
        }

        __syncthreads();                  // reads done before buf is re-targeted
    }

    // --- epilogue per 32-bit C/D layout: lane -> (n = lane&15, m0 = (lane>>4)*8)
    const int cn = lane & 15;
    const int m0 = (lane >> 4) * 8;
    #pragma unroll
    for (int i = 0; i < 2; ++i) {
        #pragma unroll
        for (int j = 0; j < 2; ++j) {
            const int col = tileN + waveN + j * 16 + cn;
            const float bv = bias[col];
            #pragma unroll
            for (int r = 0; r < 8; ++r) {
                const int row = tileM + waveM + i * 16 + m0 + r;
                if (row < M) {
                    float v = acc[i][j][r] + bv;
                    if (RELU) v = fmaxf(v, 0.0f);
                    if (OUT_BF16)
                        ((unsigned short*)Cout)[(size_t)row * N + col] = f2bf(v);
                    else
                        ((float*)Cout)[(size_t)row * N + col] = v;
                }
            }
        }
    }
}

// ---------------------------------------------------------------------------
// Host orchestration
// ---------------------------------------------------------------------------
extern "C" void kernel_launch(void* const* d_in, const int* in_sizes, int n_in,
                              void* d_out, int out_size, void* d_ws, size_t ws_size,
                              hipStream_t stream) {
    const int F = 512;   // IN_FEATS
    const int H = 512;   // N_HIDDEN
    const int C = 128;   // N_CLASSES
    const int M = in_sizes[0] / F;   // N_NODES
    const int E = in_sizes[1];       // N_EDGES

    const float* x   = (const float*)d_in[0];
    const int*   src = (const int*)d_in[1];
    const int*   dst = (const int*)d_in[2];
    const float* Ws1 = (const float*)d_in[3];
    const float* Wn1 = (const float*)d_in[4];
    const float* b1  = (const float*)d_in[5];
    const float* Ws2 = (const float*)d_in[6];
    const float* Wn2 = (const float*)d_in[7];
    const float* b2  = (const float*)d_in[8];
    float* out = (float*)d_out;

    // workspace carve-out (256B aligned)
    char* base = (char*)d_ws;
    size_t off = 0;
    auto carve = [&](size_t bytes) -> char* {
        char* p = base + off;
        off += (bytes + 255) & ~(size_t)255;
        return p;
    };
    float*          deg   = (float*)carve((size_t)M * 4);
    unsigned short* xb    = (unsigned short*)carve((size_t)M * F * 2);
    float*          Agg   = (float*)carve((size_t)M * H * 4);          // reused layer 2
    unsigned short* Aggb  = (unsigned short*)carve((size_t)M * H * 2); // reused layer 2
    unsigned short* h1b   = (unsigned short*)carve((size_t)M * H * 2);
    unsigned short* ws1t  = (unsigned short*)carve((size_t)F * H * 2); // [H][F]
    unsigned short* wn1t  = (unsigned short*)carve((size_t)F * H * 2); // [H][F]
    unsigned short* ws2t  = (unsigned short*)carve((size_t)H * C * 2); // [C][H]
    unsigned short* wn2t  = (unsigned short*)carve((size_t)H * C * 2); // [C][H]
    (void)ws_size; (void)n_in; (void)out_size;

    const int TB = 256;
    auto blks = [&](long long n) { return (unsigned)((n + TB - 1) / TB); };

    // degrees
    zero_f32_kernel<<<blks(M), TB, 0, stream>>>(deg, M);
    degree_kernel<<<blks(E), TB, 0, stream>>>(dst, deg, E);

    // bf16 conversion of x; transpose+convert of all weights to [N][K]
    convert_f32_bf16_kernel<<<blks((long long)M * F), TB, 0, stream>>>(x, xb, (long long)M * F);
    transpose_f32_bf16_kernel<<<blks((long long)F * H), TB, 0, stream>>>(Ws1, ws1t, F, H);
    transpose_f32_bf16_kernel<<<blks((long long)F * H), TB, 0, stream>>>(Wn1, wn1t, F, H);
    transpose_f32_bf16_kernel<<<blks((long long)H * C), TB, 0, stream>>>(Ws2, ws2t, H, C);
    transpose_f32_bf16_kernel<<<blks((long long)H * C), TB, 0, stream>>>(Wn2, wn2t, H, C);

    // ---- layer 1 aggregation: Agg = segment_sum(x[src]) ; Aggb = bf16(Agg/deg)
    zero_f32_kernel<<<blks((long long)M * F), TB, 0, stream>>>(Agg, (long long)M * F);
    scatter_add_f32_kernel<<<blks((long long)E * 32), TB, 0, stream>>>(x, src, dst, Agg, E, F);
    normalize_bf16_kernel<<<blks((long long)M * F), TB, 0, stream>>>(Agg, deg, Aggb, M, F);

    // ---- layer 1 fused GEMM: h1 = relu(x@Ws1 + mean@Wn1 + b1)  -> bf16
    {
        dim3 grid(H / TN, (M + TM - 1) / TM);
        sage_gemm_kernel<true, true><<<grid, 128, 0, stream>>>(
            xb, ws1t, Aggb, wn1t, b1, h1b, M, H, F);
    }

    // ---- layer 2 aggregation over h1
    zero_f32_kernel<<<blks((long long)M * H), TB, 0, stream>>>(Agg, (long long)M * H);
    scatter_add_bf16_kernel<<<blks((long long)E * 32), TB, 0, stream>>>(h1b, src, dst, Agg, E, H);
    normalize_bf16_kernel<<<blks((long long)M * H), TB, 0, stream>>>(Agg, deg, Aggb, M, H);

    // ---- layer 2 fused GEMM: out = h1@Ws2 + mean@Wn2 + b2  -> f32
    {
        dim3 grid(C / TN, (M + TM - 1) / TM);
        sage_gemm_kernel<false, false><<<grid, 128, 0, stream>>>(
            h1b, ws2t, Aggb, wn2t, b2, out, M, C, H);
    }
}
